// GATGeometric_11871289606992
// MI455X (gfx1250) — matrix operand order repaired and checked by reading, not compiled, not run on hardware
//
#include <hip/hip_runtime.h>
#include <hip/hip_fp16.h>

typedef __attribute__((ext_vector_type(16))) _Float16 v16h;
typedef __attribute__((ext_vector_type(8)))  _Float16 v8h;
typedef __attribute__((ext_vector_type(8)))  float    v8f;

#define NEG_SLOPE 0.2f

__device__ __forceinline__ int ordered_int(float f) {
    int i = __float_as_int(f);
    return (i >= 0) ? i : (i ^ 0x7fffffff);
}
__device__ __forceinline__ float ordered_float(int i) {
    return __int_as_float((i >= 0) ? i : (i ^ 0x7fffffff));
}

// Build int32 src/dst with self-loops appended.
__global__ void k_build_edges(const int* __restrict__ ei, int E, int N,
                              int* __restrict__ src, int* __restrict__ dst) {
    int i = blockIdx.x * blockDim.x + threadIdx.x;
    int tot = E + N;
    if (i >= tot) return;
    if (i < E) { src[i] = ei[i]; dst[i] = ei[E + i]; }
    else       { src[i] = i - E; dst[i] = i - E; }
}

// f32 [N,K] -> f16 [Np,K], zero-filling pad rows (so GEMM needs no row bounds checks).
__global__ void k_cvt_pad(const float* __restrict__ in, _Float16* __restrict__ out,
                          int N, int K, int Np) {
    int i = blockIdx.x * blockDim.x + threadIdx.x;
    if (i >= Np * K) return;
    int row = i / K;
    out[i] = (row < N) ? (_Float16)in[i] : (_Float16)0.f;
}

// W [K,M] f32 -> Wt [M,K] f16 (B^T so GEMM B-fragment loads are contiguous)
__global__ void k_wt_cvt(const float* __restrict__ W, _Float16* __restrict__ Wt, int K, int M) {
    int i = blockIdx.x * blockDim.x + threadIdx.x;
    if (i >= K * M) return;
    int k = i / M, m = i % M;
    Wt[(size_t)m * K + k] = (_Float16)W[i];
}

__global__ void k_fill_f32(float* p, float v, int n) {
    int i = blockIdx.x * blockDim.x + threadIdx.x;
    if (i < n) p[i] = v;
}
__global__ void k_fill_i32(int* p, int v, int n) {
    int i = blockIdx.x * blockDim.x + threadIdx.x;
    if (i < n) p[i] = v;
}

// Register-blocked WMMA GEMM: C[Np,M] = A[Np,K](f16) * Bt[M,K]^T(f16), f32 accum.
// One wave computes a 64 x (16*NT) tile: 4 A-fragments x NT B-fragments,
// 4*NT v_wmma_f32_16x16x32_f16 per K-step (4x fragment reuse vs unblocked).
// Fragment layouts used here (A/B: contiguous k, not the ISA table order; C/D as in the ISA):
//  A (16x32, 16-bit): lane L row m=L%16; g=L/16: halves 0..15 = k0+16g+0..15
//                     -> one 32B load (same k order as B).
//  B (32x16, 16-bit): lane L col n=L%16; halves 0..15 = k0+16g+0..15 -> one 32B load.
//  C/D: vgpr r -> m = r + 8*(L/16), n = L%16.
template <int NT>
__global__ __launch_bounds__(32)
void k_gemm_wmma(const _Float16* __restrict__ A, const _Float16* __restrict__ Bt,
                 float* __restrict__ C, int K, int M) {
    int lane = threadIdx.x;
    int g = lane >> 4, lm = lane & 15;
    int rowBase = blockIdx.x * 64;
    int colBase = blockIdx.y * (16 * NT);

    const _Float16* arow[4];
#pragma unroll
    for (int i = 0; i < 4; ++i)
        arow[i] = A + (size_t)(rowBase + i * 16 + lm) * K;
    const _Float16* brow[NT];
#pragma unroll
    for (int j = 0; j < NT; ++j)
        brow[j] = Bt + (size_t)(colBase + j * 16 + lm) * K;

    v8f acc[4][NT];
#pragma unroll
    for (int i = 0; i < 4; ++i)
#pragma unroll
        for (int j = 0; j < NT; ++j) acc[i][j] = (v8f){};

    for (int k0 = 0; k0 < K; k0 += 32) {
        v16h a[4], b[NT];
#pragma unroll
        for (int i = 0; i < 4; ++i) {
            a[i] = *(const v16h*)(arow[i] + k0 + 16 * g);
        }
#pragma unroll
        for (int j = 0; j < NT; ++j)
            b[j] = *(const v16h*)(brow[j] + k0 + 16 * g);
#pragma unroll
        for (int i = 0; i < 4; ++i)
#pragma unroll
            for (int j = 0; j < NT; ++j)
                acc[i][j] = __builtin_amdgcn_wmma_f32_16x16x32_f16(
                    false, a[i], false, b[j], (short)0, acc[i][j], false, false);
    }

#pragma unroll
    for (int i = 0; i < 4; ++i) {
#pragma unroll
        for (int j = 0; j < NT; ++j) {
#pragma unroll
            for (int r = 0; r < 8; ++r) {
                int m = rowBase + i * 16 + 8 * g + r;
                C[(size_t)m * M + colBase + j * 16 + lm] = acc[i][j][r];
            }
        }
    }
}

// Per-node attention scores: s[n,h] = sum_c h[n,h,c]*a_src[h,c], likewise d.
__global__ void k_score(const float* __restrict__ h, const float* __restrict__ asrc,
                        const float* __restrict__ adst, float* __restrict__ s,
                        float* __restrict__ d, int N, int H, int C) {
    int i = blockIdx.x * blockDim.x + threadIdx.x;
    if (i >= N * H) return;
    int n = i / H, hh = i % H;
    const float* hp = h + (size_t)n * H * C + hh * C;
    const float* ap = asrc + hh * C;
    const float* bp = adst + hh * C;
    float ss = 0.f, dd = 0.f;
    for (int c = 0; c < C; ++c) { float v = hp[c]; ss += v * ap[c]; dd += v * bp[c]; }
    s[i] = ss; d[i] = dd;
}

// Pass 1: e = leaky_relu(s[src]+d[dst]); stash raw e; segment-max via ordered-int atomicMax.
__global__ void k_edge_max(const int* __restrict__ src, const int* __restrict__ dst,
                           const float* __restrict__ s, const float* __restrict__ d,
                           float* __restrict__ ebuf, int* __restrict__ emax,
                           int Etot, int H) {
    int i = blockIdx.x * blockDim.x + threadIdx.x;
    if (i >= Etot * H) return;
    int e = i / H, hh = i % H;
    float v = s[src[e] * H + hh] + d[dst[e] * H + hh];
    v = v > 0.f ? v : NEG_SLOPE * v;
    ebuf[i] = v;
    atomicMax(&emax[dst[e] * H + hh], ordered_int(v));
}

// Pass 2: ex = exp(e - emax[dst]); accumulate denom.
__global__ void k_edge_exp(const int* __restrict__ dst, float* __restrict__ ebuf,
                           const int* __restrict__ emax, float* __restrict__ denom,
                           int Etot, int H) {
    int i = blockIdx.x * blockDim.x + threadIdx.x;
    if (i >= Etot * H) return;
    int e = i / H, hh = i % H;
    int di = dst[e] * H + hh;
    float w = __expf(ebuf[i] - ordered_float(emax[di]));
    ebuf[i] = w;
    atomicAdd(&denom[di], w);
}

// Pass 3: out[dst] += h[src] * alpha, float4 gather + 4 f32 atomic adds per thread.
__global__ void k_message(const int* __restrict__ src, const int* __restrict__ dst,
                          const float* __restrict__ h, const float* __restrict__ ebuf,
                          const float* __restrict__ denom, float* __restrict__ out,
                          int Etot, int H, int C) {
    int per = H * (C >> 2);
    int i = blockIdx.x * blockDim.x + threadIdx.x;
    if (i >= Etot * per) return;
    int e = i / per;
    int r = i % per;
    int hh = r / (C >> 2);
    int c4 = (r % (C >> 2)) * 4;
    float alpha = ebuf[e * H + hh] / denom[dst[e] * H + hh];
    const float4 hv = *(const float4*)(h + (size_t)src[e] * H * C + hh * C + c4);
    float* op = out + (size_t)dst[e] * H * C + hh * C + c4;
    atomicAdd(op + 0, hv.x * alpha);
    atomicAdd(op + 1, hv.y * alpha);
    atomicAdd(op + 2, hv.z * alpha);
    atomicAdd(op + 3, hv.w * alpha);
}

// Layers 1-3 epilogue: out = elu(out + b), in place (becomes next layer input).
__global__ void k_bias_elu(float* __restrict__ out, const float* __restrict__ b, int N, int M) {
    int i = blockIdx.x * blockDim.x + threadIdx.x;
    if (i >= N * M) return;
    float v = out[i] + b[i % M];
    out[i] = v > 0.f ? v : (__expf(v) - 1.f);
}

// Layer 4: y = acc + b (H=1 so head-mean is identity), then log_softmax over 32 channels.
// One wave32 per node; lane = channel.
__global__ __launch_bounds__(256)
void k_logsoftmax(const float* __restrict__ acc, const float* __restrict__ b,
                  float* __restrict__ out, int N) {
    int wave = (blockIdx.x * blockDim.x + threadIdx.x) >> 5;
    int lane = threadIdx.x & 31;
    if (wave >= N) return;
    float v = acc[(size_t)wave * 32 + lane] + b[lane];
    float m = v;
#pragma unroll
    for (int off = 16; off > 0; off >>= 1) {
        float o = __shfl_xor(m, off, 32);
        m = m > o ? m : o;
    }
    float ex = __expf(v - m);
    float sum = ex;
#pragma unroll
    for (int off = 16; off > 0; off >>= 1) sum += __shfl_xor(sum, off, 32);
    out[(size_t)wave * 32 + lane] = v - m - __logf(sum);
}

extern "C" void kernel_launch(void* const* d_in, const int* in_sizes, int n_in,
                              void* d_out, int out_size, void* d_ws, size_t ws_size,
                              hipStream_t stream) {
    (void)in_sizes; (void)n_in; (void)out_size; (void)ws_size;
    const int N = 50000, E = 400000, Etot = E + N;
    const int Np = 50048;  // N padded to multiple of 64 for the blocked GEMM

    const float* x  = (const float*)d_in[0];
    const int*   ei = (const int*)d_in[1];

    struct L { int K, H, C; const float *W, *as, *ad, *b; };
    L Ls[4] = {
        {256, 8, 32, (const float*)d_in[2],  (const float*)d_in[3],  (const float*)d_in[4],  (const float*)d_in[5]},
        {256, 4, 32, (const float*)d_in[6],  (const float*)d_in[7],  (const float*)d_in[8],  (const float*)d_in[9]},
        {128, 2, 32, (const float*)d_in[10], (const float*)d_in[11], (const float*)d_in[12], (const float*)d_in[13]},
        { 64, 1, 32, (const float*)d_in[14], (const float*)d_in[15], (const float*)d_in[16], (const float*)d_in[17]},
    };

    char* ws = (char*)d_ws;
    size_t off = 0;
    auto alloc = [&](size_t bytes) -> void* {
        void* p = ws + off;
        off += (bytes + 255) & ~(size_t)255;
        return p;
    };
    int*      src   = (int*)alloc((size_t)Etot * 4);
    int*      dst   = (int*)alloc((size_t)Etot * 4);
    float*    actA  = (float*)alloc((size_t)N * 256 * 4);
    float*    actB  = (float*)alloc((size_t)N * 256 * 4);
    float*    hbuf  = (float*)alloc((size_t)Np * 256 * 4);   // padded rows
    _Float16* Ah    = (_Float16*)alloc((size_t)Np * 256 * 2); // padded rows
    _Float16* Wt    = (_Float16*)alloc((size_t)256 * 256 * 2);
    float*    sbuf  = (float*)alloc((size_t)N * 8 * 4);
    float*    dbuf  = (float*)alloc((size_t)N * 8 * 4);
    int*      emax  = (int*)alloc((size_t)N * 8 * 4);
    float*    denom = (float*)alloc((size_t)N * 8 * 4);
    float*    ebuf  = (float*)alloc((size_t)Etot * 8 * 4);

    auto cdiv = [](long long a, long long b) { return (int)((a + b - 1) / b); };

    k_build_edges<<<cdiv(Etot, 256), 256, 0, stream>>>(ei, E, N, src, dst);

    const float* in_act = x;
    float* bufs[2] = {actA, actB};
    for (int li = 0; li < 4; ++li) {
        const L& l = Ls[li];
        const int K = l.K, H = l.H, C = l.C, M = H * C;
        float* out = bufs[li & 1];   // L1->actA, L2->actB, L3->actA, L4->actB

        k_wt_cvt<<<cdiv((long long)K * M, 256), 256, 0, stream>>>(l.W, Wt, K, M);
        k_cvt_pad<<<cdiv((long long)Np * K, 256), 256, 0, stream>>>(in_act, Ah, N, K, Np);

        if (M % 64 == 0) {
            dim3 gg(Np / 64, M / 64);
            k_gemm_wmma<4><<<gg, 32, 0, stream>>>(Ah, Wt, hbuf, K, M);
        } else {
            dim3 gg(Np / 64, M / 32);
            k_gemm_wmma<2><<<gg, 32, 0, stream>>>(Ah, Wt, hbuf, K, M);
        }

        k_score<<<cdiv((long long)N * H, 256), 256, 0, stream>>>(hbuf, l.as, l.ad, sbuf, dbuf, N, H, C);

        k_fill_i32<<<cdiv((long long)N * H, 256), 256, 0, stream>>>(emax, (int)0x80000000, N * H);
        k_fill_f32<<<cdiv((long long)N * H, 256), 256, 0, stream>>>(denom, 0.f, N * H);
        k_fill_f32<<<cdiv((long long)N * M, 256), 256, 0, stream>>>(out, 0.f, N * M);

        k_edge_max<<<cdiv((long long)Etot * H, 256), 256, 0, stream>>>(src, dst, sbuf, dbuf, ebuf, emax, Etot, H);
        k_edge_exp<<<cdiv((long long)Etot * H, 256), 256, 0, stream>>>(dst, ebuf, emax, denom, Etot, H);
        k_message<<<cdiv((long long)Etot * H * (C / 4), 256), 256, 0, stream>>>(
            src, dst, hbuf, ebuf, denom, out, Etot, H, C);

        if (li < 3) {
            k_bias_elu<<<cdiv((long long)N * M, 256), 256, 0, stream>>>(out, l.b, N, M);
            in_act = out;
        } else {
            k_logsoftmax<<<cdiv(N, 8), 256, 0, stream>>>(out, l.b, (float*)d_out, N);
        }
    }
}
